// TTTLinear_37452114821640
// MI455X (gfx1250) — compile-verified, matrix operand order checked by reading, not run on hardware
//
#include <hip/hip_runtime.h>

// ---------------------------------------------------------------------------
// TTT-Linear fused pipeline for MI455X (gfx1250, wave32, WMMA bf16).
//   K1: [8192x2048] @ [2048x4096]  -> XQK (f32) + gate=gelu(XGate) (f32)
//   K2: dwconv(q,k) + per-head Z1 = XK@W1 + b1, head-LN, +XQ -> out (f32)
//   K3: row-LN over D + ln_w/ln_b, * gate -> ygate (bf16)
//   K4: [8192x2048] @ [2048x2048]  -> d_out (f32)
// GEMMs run on v_wmma_f32_16x16x32_bf16 (f32 accumulate). f32 operands are
// staged global->VGPR (b128) -> bf16 (RNE) -> LDS (b64). K4's A operand is
// already bf16, so it uses global_load_async_to_lds_b128 (ASYNCcnt) to feed
// LDS directly, overlapped with WMMA on the other buffer.
// ---------------------------------------------------------------------------

constexpr int CB  = 2;
constexpr int CN  = 4096;
constexpr int CD  = 2048;
constexpr int CNH = 32;
constexpr int CHF = 64;          // CD / CNH
constexpr int CM  = CB * CN;     // 8192 token rows
constexpr float CEPS = 1e-6f;

typedef __attribute__((ext_vector_type(16))) __bf16        v16bf;
typedef __attribute__((ext_vector_type(8)))  float         v8f;
typedef __attribute__((ext_vector_type(4)))  float         f32x4;
typedef __attribute__((ext_vector_type(4)))  unsigned int  u32x4;
typedef __attribute__((ext_vector_type(2)))  unsigned int  u32x2;

union FragU { v16bf v; u32x4 q[2]; };

__device__ __forceinline__ unsigned short cvt_bf16(float x) {
  unsigned int u = __float_as_uint(x);
  u += 0x7FFFu + ((u >> 16) & 1u);          // round to nearest even
  return (unsigned short)(u >> 16);
}
// pack two f32 -> one dword of 2x bf16 (RNE)
__device__ __forceinline__ unsigned int pack2_bf16(float lo, float hi) {
  unsigned int ul = __float_as_uint(lo);
  unsigned int uh = __float_as_uint(hi);
  ul += 0x7FFFu + ((ul >> 16) & 1u);
  uh += 0x7FFFu + ((uh >> 16) & 1u);
  return (ul >> 16) | (uh & 0xFFFF0000u);
}
__device__ __forceinline__ float bf16f(unsigned short h) {
  return __uint_as_float(((unsigned int)h) << 16);
}
__device__ __forceinline__ float gelu_exact(float x) {
  return 0.5f * x * (1.0f + erff(x * 0.7071067811865475f));
}

// Load one 16x32 bf16 WMMA operand fragment (ISA 7.12.2 16-bit A layout:
// lanes 0-15 take K {0..7,16..23}, lanes 16-31 take K {8..15,24..31}) from an
// LDS tile stored row-major [row][k] with `stride` halves per row (16B
// aligned), k-window starting at `kofs`.
__device__ __forceinline__ v16bf ld_frag(const unsigned short* tile, int row,
                                         int stride, int kofs, int lane) {
  FragU f;
  const unsigned short* p = tile + row * stride + kofs + ((lane & 16) ? 8 : 0);
  f.q[0] = *(const u32x4*)(p);
  f.q[1] = *(const u32x4*)(p + 16);
  return f.v;
}

// ---------------------------------------------------------------------------
// K1: C[8192 x 4096] = hidden[8192 x 2048] @ qkv_w[0:4096, 0:2048]^T
//     cols <  D -> XQK,  cols >= D -> gate = gelu(.)
// ---------------------------------------------------------------------------
constexpr int BM = 128, BN = 128, BK = 32;
constexpr int KP = BK + 8;  // padded halves/row: 80B stride, 16B aligned

__global__ __launch_bounds__(256)
void k1_qkv_gemm(const float* __restrict__ hid, const float* __restrict__ qkv,
                 float* __restrict__ xqk, float* __restrict__ gate) {
  __shared__ unsigned short sA[2][BM * KP];
  __shared__ unsigned short sB[2][BN * KP];
  const int tid  = threadIdx.x;
  const int lane = tid & 31;
  const int wave = tid >> 5;
  const int wm = wave >> 1;                 // 4 waves over M (32-row strips)
  const int wn = wave & 1;                  // 2 waves over N (64-col strips)
  const int m0 = blockIdx.y * BM;
  const int n0 = blockIdx.x * BN;
  constexpr int KD = CD;
  constexpr int KT = KD / BK;               // 64 k-tiles

  f32x4 ra[4], rb[4];                       // 128x32 tile = 1024 float4 / block
  v8f acc[2][4];
  const v8f vzero = {0.f, 0.f, 0.f, 0.f, 0.f, 0.f, 0.f, 0.f};
#pragma unroll
  for (int i = 0; i < 2; ++i)
#pragma unroll
    for (int j = 0; j < 4; ++j) acc[i][j] = vzero;

  auto g_load = [&](int kt) {
    const float* pa = hid + (size_t)m0 * KD + kt * BK;
    const float* pb = qkv + (size_t)n0 * KD + kt * BK;
#pragma unroll
    for (int t = 0; t < 4; ++t) {
      int c = tid + t * 256;                // float4 chunk id, 8 per row
      int r = c >> 3, k = (c & 7) * 4;
      ra[t] = *(const f32x4*)(pa + (size_t)r * KD + k);
      rb[t] = *(const f32x4*)(pb + (size_t)r * KD + k);
    }
  };
  auto s_store = [&](int buf) {
#pragma unroll
    for (int t = 0; t < 4; ++t) {
      int c = tid + t * 256;
      int r = c >> 3, k = (c & 7) * 4;
      u32x2 pa, pb;
      pa.x = pack2_bf16(ra[t].x, ra[t].y);
      pa.y = pack2_bf16(ra[t].z, ra[t].w);
      pb.x = pack2_bf16(rb[t].x, rb[t].y);
      pb.y = pack2_bf16(rb[t].z, rb[t].w);
      *(u32x2*)&sA[buf][r * KP + k] = pa;
      *(u32x2*)&sB[buf][r * KP + k] = pb;
    }
  };

  g_load(0);
  s_store(0);
  __syncthreads();

  for (int kt = 0; kt < KT; ++kt) {
    const int cur = kt & 1;
    if (kt + 2 < KT) {  // global_prefetch_b8 for the k+2 tile
      __builtin_prefetch(hid + (size_t)(m0 + (tid >> 1)) * KD + (kt + 2) * BK, 0, 1);
      __builtin_prefetch(qkv + (size_t)(n0 + (tid >> 1)) * KD + (kt + 2) * BK, 0, 1);
    }
    if (kt + 1 < KT) g_load(kt + 1);

    const unsigned short* ta = sA[cur];
    const unsigned short* tb = sB[cur];
    v16bf af[2], bfr[4];
#pragma unroll
    for (int i = 0; i < 2; ++i)
      af[i] = ld_frag(ta, wm * 32 + i * 16 + (lane & 15), KP, 0, lane);
#pragma unroll
    for (int j = 0; j < 4; ++j)
      bfr[j] = ld_frag(tb, wn * 64 + j * 16 + (lane & 15), KP, 0, lane);
#pragma unroll
    for (int i = 0; i < 2; ++i)
#pragma unroll
      for (int j = 0; j < 4; ++j)
        acc[i][j] = __builtin_amdgcn_wmma_f32_16x16x32_bf16(
            false, af[i], false, bfr[j], (short)0, acc[i][j], false, false);

    __syncthreads();
    if (kt + 1 < KT) { s_store(cur ^ 1); __syncthreads(); }
  }

  // Epilogue: per-block the 128-col strip is entirely XQK or entirely gate.
  const bool is_gate = (n0 >= CD);
  float* outp = is_gate ? gate : xqk;
  const int ncol0 = is_gate ? (n0 - CD) : n0;
#pragma unroll
  for (int i = 0; i < 2; ++i) {
#pragma unroll
    for (int j = 0; j < 4; ++j) {
      const int gc  = ncol0 + wn * 64 + j * 16 + (lane & 15);
      const int gr0 = m0 + wm * 32 + i * 16 + ((lane >> 4) << 3);
#pragma unroll
      for (int r = 0; r < 8; ++r) {
        float v = acc[i][j][r];
        if (is_gate) v = gelu_exact(v);
        outp[(size_t)(gr0 + r) * CD + gc] = v;
      }
    }
  }
}

// ---------------------------------------------------------------------------
// K2: per (b, head, 128-token block): dwconv q/k, Z1 = XK @ W1[h] + b1,
//     head-LN(ttt_norm), out = XQ + Z1n  -> outw (f32, [B,N,D] layout)
// Dynamic LDS layout (80 KB, fine on gfx1250's 320 KB/WGP):
//   [0      ) s_x  f32 [131][64]  (token halo slice; reused as s_z [128][65])
//   [33536  ) s_k  bf16 [128][72] (conv-k output, WMMA A tile)
//   [51968  ) s_q  bf16 [128][64] (conv-q output)
//   [68352  ) s_w  bf16 [64][72]  (W1[h] transposed: [e][d])
//   [77568  ) conv weights/biases, ttt norm params, b1, per-token stats
// ---------------------------------------------------------------------------
constexpr int K2_SMEM = 81920;

__global__ __launch_bounds__(256)
void k2_conv_z1(const float* __restrict__ xqk,
                const float* __restrict__ cw_q, const float* __restrict__ cb_q,
                const float* __restrict__ cw_k, const float* __restrict__ cb_k,
                const float* __restrict__ w1g,  const float* __restrict__ b1g,
                const float* __restrict__ tnw,  const float* __restrict__ tnb,
                float* __restrict__ outw) {
  extern __shared__ char smem[];
  float*          s_x  = (float*)smem;
  unsigned short* s_k  = (unsigned short*)(smem + 33536);
  unsigned short* s_q  = (unsigned short*)(smem + 51968);
  unsigned short* s_w  = (unsigned short*)(smem + 68352);
  float*          s_cw = (float*)(smem + 77568);   // [0:256)=wq, [256:512)=wk
  float*          s_cb = s_cw + 512;               // [0:64)=bq, [64:128)=bk
  float*          s_t  = s_cb + 128;               // tw[64], tb[64], b1[64]
  float*          s_st = s_t + 192;                // mu[128], rstd[128]

  const int tid  = threadIdx.x;
  const int lane = tid & 31;
  const int wave = tid >> 5;                       // one 16-token strip each
  const int n0 = blockIdx.x * 128;
  const int h  = blockIdx.y;
  const int bb = blockIdx.z;
  const int f0 = h * CHF;

  // parameters into LDS
  for (int idx = tid; idx < 512; idx += 256) {
    int grp = idx >> 8, rem = idx & 255;
    int f = rem >> 2, kk = rem & 3;
    const float* w = grp ? cw_k : cw_q;
    s_cw[idx] = w[(size_t)(f0 + f) * 4 + kk];
  }
  if (tid < CHF) {
    s_cb[tid]      = cb_q[f0 + tid];
    s_cb[64 + tid] = cb_k[f0 + tid];
    s_t[tid]       = tnw[h * CHF + tid];
    s_t[64 + tid]  = tnb[h * CHF + tid];
    s_t[128 + tid] = b1g[h * CHF + tid];
  }
  // W1[h] transposed to [e][d] so B-fragment loads are row-contiguous
  for (int idx = tid; idx < CHF * CHF; idx += 256) {
    int e = idx >> 6, d = idx & 63;
    s_w[e * 72 + d] = cvt_bf16(w1g[((size_t)h * CHF + d) * CHF + e]);
  }
  // XQK slice with 3-token causal halo (zeros before sequence start)
  for (int idx = tid; idx < 131 * CHF; idx += 256) {
    int i = idx >> 6, f = idx & 63;
    int n = n0 + i - 3;
    s_x[idx] = (n >= 0) ? xqk[((size_t)bb * CN + n) * CD + f0 + f] : 0.0f;
  }
  __syncthreads();

  // depthwise causal conv (K=4) -> XQ (bf16), XK (bf16)
  for (int idx = tid; idx < 128 * CHF; idx += 256) {
    int t = idx >> 6, f = idx & 63;
    float aq = s_cb[f], ak = s_cb[64 + f];
#pragma unroll
    for (int kk = 0; kk < 4; ++kk) {
      float xv = s_x[(t + kk) * CHF + f];
      aq = fmaf(s_cw[f * 4 + kk], xv, aq);
      ak = fmaf(s_cw[256 + f * 4 + kk], xv, ak);
    }
    s_q[t * 64 + f] = cvt_bf16(aq);
    s_k[t * 72 + f] = cvt_bf16(ak);
  }
  __syncthreads();

  // Z1 = XK @ W1 : M=16 tokens/wave, N=64, K=64 (2 WMMA k-steps)
  v8f acc[4];
  const v8f vzero = {0.f, 0.f, 0.f, 0.f, 0.f, 0.f, 0.f, 0.f};
#pragma unroll
  for (int j = 0; j < 4; ++j) acc[j] = vzero;
#pragma unroll
  for (int ks = 0; ks < 2; ++ks) {
    v16bf a = ld_frag(s_k, wave * 16 + (lane & 15), 72, ks * 32, lane);
#pragma unroll
    for (int j = 0; j < 4; ++j) {
      v16bf bm = ld_frag(s_w, j * 16 + (lane & 15), 72, ks * 32, lane);
      acc[j] = __builtin_amdgcn_wmma_f32_16x16x32_bf16(
          false, a, false, bm, (short)0, acc[j], false, false);
    }
  }

  // spill Z1+b1 to LDS (reuse s_x region, stride 65 to dodge bank conflicts)
  float* s_z = s_x;
#pragma unroll
  for (int j = 0; j < 4; ++j) {
    const int e  = j * 16 + (lane & 15);
    const int t0 = wave * 16 + ((lane >> 4) << 3);
#pragma unroll
    for (int r = 0; r < 8; ++r)
      s_z[(t0 + r) * 65 + e] = acc[j][r] + s_t[128 + e];
  }
  __syncthreads();

  // per-token LN stats over HF=64
  if (tid < 128) {
    float s = 0.f, ss = 0.f;
    for (int e = 0; e < CHF; ++e) {
      float z = s_z[tid * 65 + e];
      s += z; ss += z * z;
    }
    float mu  = s * (1.0f / CHF);
    float var = ss * (1.0f / CHF) - mu * mu;
    s_st[tid]       = mu;
    s_st[128 + tid] = rsqrtf(var + CEPS);
  }
  __syncthreads();

  // out = XQ + ttt_norm(Z1)
  for (int idx = tid; idx < 128 * CHF; idx += 256) {
    int t = idx >> 6, f = idx & 63;
    float zh = (s_z[t * 65 + f] - s_st[t]) * s_st[128 + t];
    float zn = s_t[f] * zh + s_t[64 + f];
    outw[((size_t)bb * CN + n0 + t) * CD + f0 + f] = bf16f(s_q[t * 64 + f]) + zn;
  }
}

// ---------------------------------------------------------------------------
// K3: row-LN over D=2048, scale/shift, multiply by gate, emit bf16.
// ---------------------------------------------------------------------------
__global__ __launch_bounds__(256)
void k3_ln_gate(const float* __restrict__ outw, const float* __restrict__ gate,
                const float* __restrict__ lnw,  const float* __restrict__ lnb,
                unsigned short* __restrict__ yg) {
  __shared__ float red[256];
  const int row = blockIdx.x;
  const float* x = outw + (size_t)row * CD;
  float v[8], s = 0.f, ss = 0.f;
#pragma unroll
  for (int j = 0; j < 8; ++j) {
    float t = x[threadIdx.x + j * 256];
    v[j] = t; s += t; ss += t * t;
  }
  red[threadIdx.x] = s;  __syncthreads();
  for (int o = 128; o > 0; o >>= 1) {
    if (threadIdx.x < o) red[threadIdx.x] += red[threadIdx.x + o];
    __syncthreads();
  }
  const float mean = red[0] * (1.0f / CD);
  __syncthreads();
  red[threadIdx.x] = ss; __syncthreads();
  for (int o = 128; o > 0; o >>= 1) {
    if (threadIdx.x < o) red[threadIdx.x] += red[threadIdx.x + o];
    __syncthreads();
  }
  const float var = red[0] * (1.0f / CD) - mean * mean;
  const float rs  = rsqrtf(var + CEPS);
#pragma unroll
  for (int j = 0; j < 8; ++j) {
    const int c = threadIdx.x + j * 256;
    float yn = (v[j] - mean) * rs * lnw[c] + lnb[c];
    float g  = gate[(size_t)row * CD + c];
    yg[(size_t)row * CD + c] = cvt_bf16(yn * g);
  }
}

// ---------------------------------------------------------------------------
// K4: d_out[8192 x 2048] = ygate(bf16) @ wo_w[2048 x 2048]^T   (f32 out)
// A operand is bf16 in memory: stream it straight into LDS with
// global_load_async_to_lds_b128 (ASYNCcnt), overlapped with WMMA compute.
// ---------------------------------------------------------------------------
__global__ __launch_bounds__(256)
void k4_out_gemm(const unsigned short* __restrict__ Abf,
                 const float* __restrict__ Wt, float* __restrict__ out) {
  __shared__ unsigned short sA[2][BM * KP];
  __shared__ unsigned short sB[2][BN * KP];
  const int tid  = threadIdx.x;
  const int lane = tid & 31;
  const int wave = tid >> 5;
  const int wm = wave >> 1, wn = wave & 1;
  const int m0 = blockIdx.y * BM, n0 = blockIdx.x * BN;
  constexpr int KD = CD, KT = KD / BK;

  f32x4 rb[4];
  v8f acc[2][4];
  const v8f vzero = {0.f, 0.f, 0.f, 0.f, 0.f, 0.f, 0.f, 0.f};
#pragma unroll
  for (int i = 0; i < 2; ++i)
#pragma unroll
    for (int j = 0; j < 4; ++j) acc[i][j] = vzero;

  // A tile (128 rows x 32 bf16 = 8 KB): 2 async b128 copies per thread.
  auto a_async = [&](int buf, int kt) {
#pragma unroll
    for (int j = 0; j < 2; ++j) {
      int c = tid + 256 * j;                 // 16B chunk id, 4 per row
      int r = c >> 2, kq = (c & 3) * 8;      // kq in halves
      unsigned int ldsa =
          (unsigned int)(uintptr_t)&sA[buf][r * KP + kq];
      const unsigned short* gp =
          Abf + (size_t)(m0 + r) * KD + kt * BK + kq;
      asm volatile("global_load_async_to_lds_b128 %0, %1, off"
                   :: "v"(ldsa), "v"((unsigned long long)(uintptr_t)gp)
                   : "memory");
    }
  };
  auto b_load = [&](int kt) {
    const float* pb = Wt + (size_t)n0 * KD + kt * BK;
#pragma unroll
    for (int t = 0; t < 4; ++t) {
      int c = tid + t * 256;
      int r = c >> 3, k = (c & 7) * 4;
      rb[t] = *(const f32x4*)(pb + (size_t)r * KD + k);
    }
  };
  auto b_store = [&](int buf) {
#pragma unroll
    for (int t = 0; t < 4; ++t) {
      int c = tid + t * 256;
      int r = c >> 3, k = (c & 7) * 4;
      u32x2 p;
      p.x = pack2_bf16(rb[t].x, rb[t].y);
      p.y = pack2_bf16(rb[t].z, rb[t].w);
      *(u32x2*)&sB[buf][r * KP + k] = p;
    }
  };

  a_async(0, 0);
  b_load(0);
  b_store(0);
  asm volatile("s_wait_asynccnt 0x0" ::: "memory");
  __syncthreads();

  for (int kt = 0; kt < KT; ++kt) {
    const int cur = kt & 1;
    if (kt + 2 < KT)
      __builtin_prefetch(Wt + (size_t)(n0 + (tid >> 1)) * KD + (kt + 2) * BK, 0, 1);
    if (kt + 1 < KT) {
      a_async(cur ^ 1, kt + 1);   // buffer nxt: last read at kt-1, barrier passed
      b_load(kt + 1);
    }

    const unsigned short* ta = sA[cur];
    const unsigned short* tb = sB[cur];
    v16bf af[2], bfr[4];
#pragma unroll
    for (int i = 0; i < 2; ++i)
      af[i] = ld_frag(ta, wm * 32 + i * 16 + (lane & 15), KP, 0, lane);
#pragma unroll
    for (int j = 0; j < 4; ++j)
      bfr[j] = ld_frag(tb, wn * 64 + j * 16 + (lane & 15), KP, 0, lane);
#pragma unroll
    for (int i = 0; i < 2; ++i)
#pragma unroll
      for (int j = 0; j < 4; ++j)
        acc[i][j] = __builtin_amdgcn_wmma_f32_16x16x32_bf16(
            false, af[i], false, bfr[j], (short)0, acc[i][j], false, false);

    __syncthreads();
    if (kt + 1 < KT) {
      b_store(cur ^ 1);
      asm volatile("s_wait_asynccnt 0x0" ::: "memory");
      __syncthreads();
    }
  }

#pragma unroll
  for (int i = 0; i < 2; ++i) {
#pragma unroll
    for (int j = 0; j < 4; ++j) {
      const int gc  = n0 + wn * 64 + j * 16 + (lane & 15);
      const int gr0 = m0 + wm * 32 + i * 16 + ((lane >> 4) << 3);
#pragma unroll
      for (int r = 0; r < 8; ++r)
        out[(size_t)(gr0 + r) * CD + gc] = acc[i][j][r];
    }
  }
}

// ---------------------------------------------------------------------------
// host launch
// ---------------------------------------------------------------------------
extern "C" void kernel_launch(void* const* d_in, const int* in_sizes, int n_in,
                              void* d_out, int out_size, void* d_ws, size_t ws_size,
                              hipStream_t stream) {
  (void)in_sizes; (void)n_in; (void)out_size; (void)ws_size;
  const float* hid = (const float*)d_in[0];
  const float* qkv = (const float*)d_in[1];
  const float* cqw = (const float*)d_in[2];
  const float* cqb = (const float*)d_in[3];
  const float* ckw = (const float*)d_in[4];
  const float* ckb = (const float*)d_in[5];
  const float* w1  = (const float*)d_in[6];
  const float* b1  = (const float*)d_in[7];
  const float* tnw = (const float*)d_in[8];
  const float* tnb = (const float*)d_in[9];
  const float* lnw = (const float*)d_in[10];
  const float* lnb = (const float*)d_in[11];
  const float* wo  = (const float*)d_in[12];

  // workspace layout (peak 192 MB; ygate reuses the dead XQK region)
  const size_t plane = (size_t)CM * CD * sizeof(float);   // 64 MB
  char* ws = (char*)d_ws;
  float*          xqk  = (float*)(ws);
  float*          gate = (float*)(ws + plane);
  float*          outw = (float*)(ws + 2 * plane);
  unsigned short* yg   = (unsigned short*)(ws);           // after K2, XQK dead

  (void)hipFuncSetAttribute((const void*)k2_conv_z1,
                            hipFuncAttributeMaxDynamicSharedMemorySize, K2_SMEM);

  dim3 g1((2 * CD) / BN, CM / BM);        // 32 x 64
  k1_qkv_gemm<<<g1, 256, 0, stream>>>(hid, qkv, xqk, gate);

  dim3 g2(CN / 128, CNH, CB);             // 32 x 32 x 2
  k2_conv_z1<<<g2, 256, K2_SMEM, stream>>>(xqk, cqw, cqb, ckw, ckb,
                                           w1, b1, tnw, tnb, outw);

  k3_ln_gate<<<dim3(CM), 256, 0, stream>>>(outw, gate, lnw, lnb, yg);

  dim3 g4(CD / BN, CM / BM);              // 16 x 64
  k4_out_gemm<<<g4, 256, 0, stream>>>(yg, wo, (float*)d_out);
}
